// Decoder_71502615543888
// MI455X (gfx1250) — compile-verified, matrix operand order checked by reading, not run on hardware
//
#include <hip/hip_runtime.h>
#include <hip/hip_bf16.h>

typedef __attribute__((ext_vector_type(16))) _Float16 v16h;
typedef __attribute__((ext_vector_type(8)))  float    v8f;

#define HH 28
#define WW 64
#define PIX (HH*WW)          // 1792
#define BB 16
#define NCH 631
#define KCHUNK 448           // K elements staged per LDS chunk (multiple of 64)
#define KSTRIDE 456          // LDS row stride in halfs: 912B, 16B-aligned, conflict-free phase

__device__ __forceinline__ float lrelu(float v) { return v >= 0.f ? v : 0.1f * v; }

// ---------------- quaternion -> M = K * R * Kinv (per batch) ----------------
__global__ void k_rotmat(const float* __restrict__ quat, const float* __restrict__ Km,
                         const float* __restrict__ Kinv, float* __restrict__ Mout) {
    int b = threadIdx.x;
    if (b >= BB) return;
    float qw = quat[b*4+0], qx = quat[b*4+1], qy = quat[b*4+2], qz = quat[b*4+3];
    float inv = rsqrtf(qw*qw + qx*qx + qy*qy + qz*qz);
    qw *= inv; qx *= inv; qy *= inv; qz *= inv;
    float R[9] = {1-2*(qy*qy+qz*qz), 2*(qx*qy-qw*qz), 2*(qx*qz+qw*qy),
                  2*(qx*qy+qw*qz), 1-2*(qx*qx+qz*qz), 2*(qy*qz-qw*qx),
                  2*(qx*qz-qw*qy), 2*(qy*qz+qw*qx), 1-2*(qx*qx+qy*qy)};
    float KR[9];
    for (int i = 0; i < 3; ++i)
        for (int j = 0; j < 3; ++j) {
            float s = 0.f;
            for (int k = 0; k < 3; ++k) s += Km[i*3+k] * R[k*3+j];
            KR[i*3+j] = s;
        }
    for (int i = 0; i < 3; ++i)
        for (int j = 0; j < 3; ++j) {
            float s = 0.f;
            for (int k = 0; k < 3; ++k) s += KR[i*3+k] * Kinv[k*3+j];
            Mout[b*9 + i*3 + j] = s;
        }
}

// -------- rotation flow, evaluated on the fly under 16x triangle filter, /16 --------
__global__ void k_rflow(const float* __restrict__ M, float* __restrict__ out) {
    int i = blockIdx.x * blockDim.x + threadIdx.x;
    if (i >= BB * PIX) return;
    int x = i & 63, y = (i >> 6) % HH, b = i / PIX;
    const float* m = M + b * 9;
    float cy = (y + 0.5f) * 16.f - 0.5f;
    float cx = (x + 0.5f) * 16.f - 0.5f;
    int y0 = (int)ceilf(cy - 16.f); if (y0 < 0) y0 = 0;
    int y1 = (int)floorf(cy + 16.f); if (y1 > 447) y1 = 447;
    int xa = (int)ceilf(cx - 16.f); if (xa < 0) xa = 0;
    int xb = (int)floorf(cx + 16.f); if (xb > 1023) xb = 1023;
    float su = 0.f, sv = 0.f, sw = 0.f;
    for (int Y = y0; Y <= y1; ++Y) {
        float wy = 1.f - fabsf((float)Y - cy) * 0.0625f;
        if (wy <= 0.f) continue;
        for (int X = xa; X <= xb; ++X) {
            float wx = 1.f - fabsf((float)X - cx) * 0.0625f;
            if (wx <= 0.f) continue;
            float w = wy * wx;
            float fX = (float)X, fY = (float)Y;
            float rd = 1.f / (m[6]*fX + m[7]*fY + m[8]);
            su += w * ((m[0]*fX + m[1]*fY + m[2]) * rd - fX);
            sv += w * ((m[3]*fX + m[4]*fY + m[5]) * rd - fY);
            sw += w;
        }
    }
    float inv = 1.f / (sw * 16.f);   // normalize triangle filter, then /SCALE
    out[((size_t)b*2    )*PIX + y*WW + x] = su * inv;
    out[((size_t)b*2 + 1)*PIX + y*WW + x] = sv * inv;
}

// ---------------- transposed conv 4x4 stride 2 (pad 2), Cout = 2 ----------------
__global__ void k_deconv2(const float* __restrict__ x, const float* __restrict__ w,
                          const float* __restrict__ bias, float* __restrict__ out, int Cin) {
    int i = blockIdx.x * blockDim.x + threadIdx.x;
    if (i >= BB * 2 * PIX) return;
    int xx = i & 63, y = (i >> 6) % HH, co = (i / PIX) & 1, b = i / (2 * PIX);
    float s = bias[co];
    for (int ky = 0; ky < 4; ++ky) {
        int ty = y + ky - 2;
        if (ty & 1) continue;
        int iy = ty >> 1;
        if (iy < 0 || iy >= 14) continue;
        for (int kx = 0; kx < 4; ++kx) {
            int tx = xx + kx - 2;
            if (tx & 1) continue;
            int ix = tx >> 1;
            if (ix < 0 || ix >= 32) continue;
            int fy = 3 - ky, fx = 3 - kx;
            const float* xp = x + (size_t)b * Cin * 448 + iy * 32 + ix;
            const float* wp = w + (size_t)co * 16 + fy * 4 + fx;   // w[ci][co][fy][fx]
            for (int ci = 0; ci < Cin; ++ci)
                s += xp[(size_t)ci * 448] * wp[(size_t)ci * 32];
        }
    }
    out[i] = s;
}

// ---------------- bilinear warp with per-corner validity mask ----------------
__global__ void k_warp(const float* __restrict__ src, const float* __restrict__ flow,
                       float flowScale, float* __restrict__ dst, int C) {
    int i = blockIdx.x * blockDim.x + threadIdx.x;
    if (i >= BB * C * PIX) return;
    int x = i & 63, y = (i >> 6) % HH;
    int c = (i / PIX) % C, b = i / (PIX * C);
    size_t fb = (size_t)b * 2 * PIX + y * WW + x;
    float X = (float)x + flow[fb] * flowScale;
    float Y = (float)y + flow[fb + PIX] * flowScale;
    float x0f = floorf(X), y0f = floorf(Y);
    float wx = X - x0f, wy = Y - y0f;
    const float* plane = src + ((size_t)b * C + c) * PIX;
    auto g = [&](float yi, float xi) -> float {
        float valid = (yi >= 0.f && yi <= (float)(HH-1) && xi >= 0.f && xi <= (float)(WW-1)) ? 1.f : 0.f;
        int yc = (int)fminf(fmaxf(yi, 0.f), (float)(HH-1));
        int xc = (int)fminf(fmaxf(xi, 0.f), (float)(WW-1));
        return plane[yc * WW + xc] * valid;
    };
    dst[i] = g(y0f, x0f)       * (1.f-wy)*(1.f-wx) + g(y0f, x0f+1.f)     * (1.f-wy)*wx
           + g(y0f+1.f, x0f)   * wy*(1.f-wx)       + g(y0f+1.f, x0f+1.f) * wy*wx;
}

// ------------- correlation (81ch cost volume) + lrelu -> feat ch 448..528 -------------
__global__ void k_corr(const float* __restrict__ one, const float* __restrict__ two,
                       float* __restrict__ out32, _Float16* __restrict__ out16) {
    int i = blockIdx.x * blockDim.x + threadIdx.x;
    if (i >= BB * 81 * PIX) return;
    int x = i & 63, y = (i >> 6) % HH, d = (i / PIX) % 81, b = i / (PIX * 81);
    int dy = d / 9 - 4, dx = d % 9 - 4;
    int yy = y + dy, xx = x + dx;
    float s = 0.f;
    if (yy >= 0 && yy < HH && xx >= 0 && xx < WW) {
        const float* p1 = one + (size_t)b * 96 * PIX + y * WW + x;
        const float* p2 = two + (size_t)b * 96 * PIX + yy * WW + xx;
        for (int c = 0; c < 96; ++c) s += p1[(size_t)c * PIX] * p2[(size_t)c * PIX];
    }
    s = lrelu(s * (1.f / 96.f));
    size_t o = (((size_t)b * NCH + 448 + d) * HH + y) * WW + x;
    out32[o] = s;
    out16[o] = (_Float16)s;
}

// -------- assemble remaining base channels 529..630: Rflow(2)+tenOne(96)+tenFlow(2)+tenFeatUp(2) --------
__global__ void k_base(const float* __restrict__ rflow, const float* __restrict__ one,
                       const float* __restrict__ tflow, const float* __restrict__ tfup,
                       float* __restrict__ out32, _Float16* __restrict__ out16) {
    int i = blockIdx.x * blockDim.x + threadIdx.x;
    if (i >= BB * 102 * PIX) return;
    int x = i & 63, y = (i >> 6) % HH, j = (i / PIX) % 102, b = i / (PIX * 102);
    int pix = y * WW + x;
    float v;
    if (j < 2)        v = rflow[((size_t)b*2 + j)        * PIX + pix];
    else if (j < 98)  v = one  [((size_t)b*96 + (j-2))   * PIX + pix];
    else if (j < 100) v = tflow[((size_t)b*2 + (j-98))   * PIX + pix];
    else              v = tfup [((size_t)b*2 + (j-100))  * PIX + pix];
    size_t o = (((size_t)b * NCH + 529 + j) * HH + y) * WW + x;
    out32[o] = v;
    out16[o] = (_Float16)v;
}

// -------- pack f32 OIHW weights into f16 WMMA B-fragments: [ktile][ntile][lane][slot] --------
__global__ void k_packw(const float* __restrict__ w, _Float16* __restrict__ pw,
                        int Cin, int Cout, int ntiles, int K, int ktiles) {
    size_t e = (size_t)blockIdx.x * blockDim.x + threadIdx.x;
    size_t total = (size_t)ktiles * ntiles * 512;
    if (e >= total) return;
    int slot  = (int)(e & 15);
    int lane  = (int)((e >> 4) & 31);
    size_t fr = e >> 9;
    int nt = (int)(fr % ntiles);
    int kt = (int)(fr / ntiles);
    int n  = nt * 16 + (lane & 15);
    int half_ = lane >> 4;
    int v_ = slot >> 1, p = slot & 1;
    int k = kt * 32 + ((v_ < 4) ? 0 : 16) + ((v_ & 3) * 2 + p) + half_ * 8;
    float val = 0.f;
    if (n < Cout && k < K) {
        int ci = k / 9, rem = k % 9, ky = rem / 3, kx = rem % 3;
        val = w[(((size_t)n * Cin + ci) * 3 + ky) * 3 + kx];
    }
    pw[e] = (_Float16)val;
}

// ---- implicit-GEMM 3x3 conv: 64 pixels (4 M-subtiles) x 16 Cout per wave ----
// K loop unrolled x2 with double-buffered B fragments (bf/bg), unconditional
// clamped prefetch (prefetch distance = 2 fragments, carried across chunks).
__global__ __launch_bounds__(256) void k_conv(
    const _Float16* __restrict__ feat, const _Float16* __restrict__ pw,
    const float* __restrict__ bias, float* __restrict__ out32, _Float16* __restrict__ out16,
    int chin0, int Cout, int ntiles, int chout0, int K, int Kpad, int flowOut) {
    __shared__ _Float16 ldsA[64 * KSTRIDE];   // ~57 KB
    int t = blockIdx.x;                 // 0 .. BB*HH-1 : one full image row per block
    int b = t / HH, y = t % HH;
    int tid = threadIdx.x;
    int lane = tid & 31, wave = tid >> 5;
    int half_ = lane >> 4, lm = lane & 15;
    bool active = wave < ntiles;

    auto loadB = [&](int kt) -> v16h {
        return *(const v16h*)(pw + (((size_t)kt * ntiles + wave) * 32 + lane) * 16);
    };
    auto loadA = [&](int row, int kk) -> v16h {
        const _Float16* r = &ldsA[row * KSTRIDE + kk];
        v16h a;
#pragma unroll
        for (int sl = 0; sl < 16; ++sl) {
            int v_ = sl >> 1, p = sl & 1;
            int ko = ((v_ < 4) ? 0 : 16) + ((v_ & 3) * 2 + p) + half_ * 8;
            a[sl] = r[ko];
        }
        return a;
    };

    int nkt = Kpad >> 5;                // Kpad multiple of 32; odd (179) for the last layer
    v8f acc0 = {}, acc1 = {}, acc2 = {}, acc3 = {};
    v16h bf = {}, bg = {};
    if (active) {
        bf = loadB(0);
        bg = loadB(1 < nkt ? 1 : 0);
    }
    int kt = 0;

    for (int kbase = 0; kbase < Kpad; kbase += KCHUNK) {
        int kfill = Kpad - kbase; if (kfill > KCHUNK) kfill = KCHUNK;  // multiple of 32
        // cooperative im2col into LDS, k-outer / m-inner for coalesced feat reads
        for (int idx = tid; idx < kfill * 64; idx += 256) {
            int kk = idx >> 6, m = idx & 63;
            int k = kbase + kk;
            _Float16 v = (_Float16)0.f;
            if (k < K) {
                int ci = k / 9, rem = k % 9;
                int ky = rem / 3, kx = rem % 3;
                int yy = y + ky - 1, xx = m + kx - 1;
                if (yy >= 0 && yy < HH && xx >= 0 && xx < WW)
                    v = feat[(((size_t)b * NCH + chin0 + ci) * HH + yy) * WW + xx];
            }
            ldsA[m * KSTRIDE + kk] = v;
        }
        __syncthreads();
        if (active) {
            for (int kk = 0; kk < kfill; kk += 64) {
                // ---- group 0: consume bf, refill bf for kt+2 ----
                {
                    v16h a0 = loadA(lm,      kk), a1 = loadA(16 + lm, kk);
                    v16h a2 = loadA(32 + lm, kk), a3 = loadA(48 + lm, kk);
                    acc0 = __builtin_amdgcn_wmma_f32_16x16x32_f16(false, a0, false, bf, (short)0, acc0, false, false);
                    acc1 = __builtin_amdgcn_wmma_f32_16x16x32_f16(false, a1, false, bf, (short)0, acc1, false, false);
                    acc2 = __builtin_amdgcn_wmma_f32_16x16x32_f16(false, a2, false, bf, (short)0, acc2, false, false);
                    acc3 = __builtin_amdgcn_wmma_f32_16x16x32_f16(false, a3, false, bf, (short)0, acc3, false, false);
                }
                bf = loadB(kt + 2 < nkt ? kt + 2 : nkt - 1);
                // ---- group 1: consume bg, refill bg for kt+3 ----
                {
                    v16h a0 = loadA(lm,      kk + 32), a1 = loadA(16 + lm, kk + 32);
                    v16h a2 = loadA(32 + lm, kk + 32), a3 = loadA(48 + lm, kk + 32);
                    acc0 = __builtin_amdgcn_wmma_f32_16x16x32_f16(false, a0, false, bg, (short)0, acc0, false, false);
                    acc1 = __builtin_amdgcn_wmma_f32_16x16x32_f16(false, a1, false, bg, (short)0, acc1, false, false);
                    acc2 = __builtin_amdgcn_wmma_f32_16x16x32_f16(false, a2, false, bg, (short)0, acc2, false, false);
                    acc3 = __builtin_amdgcn_wmma_f32_16x16x32_f16(false, a3, false, bg, (short)0, acc3, false, false);
                }
                bg = loadB(kt + 3 < nkt ? kt + 3 : nkt - 1);
                kt += 2;
            }
        }
        __syncthreads();
    }
    if (active) {
        int n = wave * 16 + lm;
        if (n < Cout) {
            float bv = bias[n];
#pragma unroll
            for (int mt = 0; mt < 4; ++mt) {
                v8f acc = (mt == 0) ? acc0 : (mt == 1) ? acc1 : (mt == 2) ? acc2 : acc3;
#pragma unroll
                for (int r = 0; r < 8; ++r) {
                    int m = r + 8 * half_;                 // D layout: VGPR r -> M = r + 8*half
                    float val = lrelu(acc[r] + bv);
                    int xx = mt * 16 + m;
                    if (flowOut) {
                        out32[(((size_t)b * 2 + n) * HH + y) * WW + xx] = val;
                    } else {
                        size_t o = (((size_t)b * NCH + chout0 + n) * HH + y) * WW + xx;
                        out32[o] = val;
                        out16[o] = (_Float16)val;
                    }
                }
            }
        }
    }
}

extern "C" void kernel_launch(void* const* d_in, const int* in_sizes, int n_in,
                              void* d_out, int out_size, void* d_ws, size_t ws_size,
                              hipStream_t stream) {
    (void)in_sizes; (void)n_in; (void)out_size; (void)ws_size;
    const float* tenOne   = (const float*)d_in[0];
    const float* tenTwo   = (const float*)d_in[1];
    const float* prevFlow = (const float*)d_in[2];
    const float* prevFeat = (const float*)d_in[3];
    const float* quat     = (const float*)d_in[4];
    const float* Km       = (const float*)d_in[5];
    const float* Kinv     = (const float*)d_in[6];
    const float* upflow_w = (const float*)d_in[7];
    const float* upflow_b = (const float*)d_in[8];
    const float* upfeat_w = (const float*)d_in[9];
    const float* upfeat_b = (const float*)d_in[10];

    float* out = (float*)d_out;
    float* flowOut32 = out;                    // (16,2,28,64)
    float* featOut32 = out + (size_t)BB*2*PIX; // (16,631,28,64)

    char* ws = (char*)d_ws;
    size_t off = 0;
    auto take = [&](size_t bytes) -> void* {
        void* p = ws + off;
        off = (off + bytes + 255) & ~(size_t)255;
        return p;
    };
    float*     Mbuf    = (float*)take((size_t)BB*9*4);
    float*     rflow   = (float*)take((size_t)BB*2*PIX*4);
    float*     tflow   = (float*)take((size_t)BB*2*PIX*4);
    float*     tfup    = (float*)take((size_t)BB*2*PIX*4);
    float*     two_w   = (float*)take((size_t)BB*96*PIX*4);
    float*     warped2 = (float*)take((size_t)BB*96*PIX*4);
    _Float16*  featF16 = (_Float16*)take((size_t)BB*NCH*PIX*2);
    _Float16*  packedW = (_Float16*)take((size_t)1600000*2);

    k_rotmat<<<1, 32, 0, stream>>>(quat, Km, Kinv, Mbuf);
    {
        int tot = BB * PIX;
        k_rflow<<<(tot + 127) / 128, 128, 0, stream>>>(Mbuf, rflow);
    }
    {
        int tot = BB * 2 * PIX;
        k_deconv2<<<(tot + 255) / 256, 256, 0, stream>>>(prevFlow, upflow_w, upflow_b, tflow, 2);
        k_deconv2<<<(tot + 255) / 256, 256, 0, stream>>>(prevFeat, upfeat_w, upfeat_b, tfup, 663);
    }
    {
        int tot = BB * 96 * PIX;
        k_warp<<<(tot + 255) / 256, 256, 0, stream>>>(tenTwo, rflow, 1.00f, two_w,   96);
        k_warp<<<(tot + 255) / 256, 256, 0, stream>>>(two_w,  tflow, 1.25f, warped2, 96);
    }
    {
        int tot = BB * 81 * PIX;
        k_corr<<<(tot + 255) / 256, 256, 0, stream>>>(tenOne, warped2, featOut32, featF16);
        tot = BB * 102 * PIX;
        k_base<<<(tot + 255) / 256, 256, 0, stream>>>(rflow, tenOne, tflow, tfup, featOut32, featF16);
    }

    // layer table: chin0, Cin, Cout, ntiles, chout0, K, Kpad (multiple of 32), wIdx, bIdx
    struct L { int chin0, Cin, Cout, nt, chout0, K, Kpad, wIdx, bIdx; };
    const L layers[6] = {
        {448, 183, 128, 8, 320, 1647, 1664, 11, 12},
        {320, 311, 128, 8, 192, 2799, 2816, 13, 14},
        {192, 439,  96, 6,  96, 3951, 3968, 15, 16},
        { 96, 535,  64, 4,  32, 4815, 4864, 17, 18},
        { 32, 599,  32, 2,   0, 5391, 5440, 19, 20},
        {  0, 631,   2, 1,   0, 5679, 5728, 21, 22},
    };

    size_t pwOff = 0;
    for (int l = 0; l < 6; ++l) {
        const L& a = layers[l];
        int ktiles = a.Kpad / 32;
        size_t tot = (size_t)ktiles * a.nt * 512;
        k_packw<<<(unsigned)((tot + 255) / 256), 256, 0, stream>>>(
            (const float*)d_in[a.wIdx], packedW + pwOff, a.Cin, a.Cout, a.nt, a.K, ktiles);
        pwOff += tot;
    }
    pwOff = 0;
    for (int l = 0; l < 6; ++l) {
        const L& a = layers[l];
        int ktiles = a.Kpad / 32;
        size_t tot = (size_t)ktiles * a.nt * 512;
        k_conv<<<BB * HH, 256, 0, stream>>>(
            featF16, packedW + pwOff, (const float*)d_in[a.bIdx],
            (l == 5) ? flowOut32 : featOut32, featF16,
            a.chin0, a.Cout, a.nt, a.chout0, a.K, a.Kpad, (l == 5) ? 1 : 0);
        pwOff += tot;
    }
}
